// LinearCryptoLite_35321811042585
// MI455X (gfx1250) — matrix operand order repaired and checked by reading, not compiled, not run on hardware
//
#include <hip/hip_runtime.h>
#include <hip/hip_bf16.h>
#include <math.h>

// Problem dims (fixed by reference)
#define B_ 32
#define F_ 64
#define L_ 8192
#define P_ 96
#define H_ 8

typedef __attribute__((ext_vector_type(16))) _Float16 v16h;
typedef __attribute__((ext_vector_type(8)))  float    v8f;

__device__ __forceinline__ v8f wmma16(v16h a, v16h b, v8f c) {
  // D = A(16x32 f16) x B(32x16 f16) + C(16x16 f32)
  return __builtin_amdgcn_wmma_f32_16x16x32_f16(false, a, false, b, (short)0, c,
                                                false, false);
}

// A-matrix 16x32 f16 fragment: lane group g = lane>>4, m = lane&15,
// VGPR v holds f16 pair at k = 2*v + 16*g.  Per lane-group the
// 8 pairs form one 32B-contiguous run -> compiler emits ds_load_b128 / b64.
template <typename T>
__device__ __forceinline__ v16h loadA16x32(const T* base, int rowStride, int lane,
                                           int kbase) {
  int g = (lane >> 4) & 1;
  int m = lane & 15;
  v16h a;
#pragma unroll
  for (int v = 0; v < 8; ++v) {
    int k = kbase + (v << 1) + (g << 4);
    a[2 * v]     = (_Float16)base[m * rowStride + k];
    a[2 * v + 1] = (_Float16)base[m * rowStride + k + 1];
  }
  return a;
}

// B-matrix 32x16 f16 fragment, B(k,n) = W[(nbase+n)*64 + k]  (i.e. H @ W^T)
template <typename T>
__device__ __forceinline__ v16h loadB_WT(const T* W, int lane, int kbase, int nbase) {
  int g = (lane >> 4) & 1;
  int n = (lane & 15) + nbase;
  v16h b;
#pragma unroll
  for (int v = 0; v < 8; ++v) {
    int k = kbase + (v << 1) + (g << 4);
    b[2 * v]     = (_Float16)W[n * 64 + k];
    b[2 * v + 1] = (_Float16)W[n * 64 + k + 1];
  }
  return b;
}

// B-matrix 32x16 f16 fragment from row-major M[k][n] with row stride rs.
template <typename T>
__device__ __forceinline__ v16h loadB_RM(const T* M, int rs, int lane, int kbase,
                                         int nbase) {
  int g = (lane >> 4) & 1;
  int n = (lane & 15) + nbase;
  v16h b;
#pragma unroll
  for (int v = 0; v < 8; ++v) {
    int k = kbase + (v << 1) + (g << 4);
    b[2 * v]     = (_Float16)M[(size_t)k * rs + n];
    b[2 * v + 1] = (_Float16)M[(size_t)(k + 1) * rs + n];
  }
  return b;
}

// elu(x)+1, branchless: native v_exp_f32 + cndmask (no exec-mask juggling).
__device__ __forceinline__ float phi_f(float v) {
  float e = __expf(fminf(v, 0.f));
  return v > 0.f ? v + 1.f : e;
}

// ---------------- Kernel 1: RevIN stats (mean, unbiased std + eps) -------------
__global__ void k_stats(const float* __restrict__ x, float* __restrict__ mean,
                        float* __restrict__ stde) {
  __shared__ float s1[256], s2[256];
  int bf = blockIdx.x;  // b*F + f
  const float* xp = x + (size_t)bf * L_;
  float s = 0.f, q = 0.f;
  for (int i = threadIdx.x; i < L_; i += 256) {
    float v = xp[i];
    s += v; q += v * v;
  }
  s1[threadIdx.x] = s; s2[threadIdx.x] = q;
  __syncthreads();
  for (int st = 128; st > 0; st >>= 1) {
    if ((int)threadIdx.x < st) {
      s1[threadIdx.x] += s1[threadIdx.x + st];
      s2[threadIdx.x] += s2[threadIdx.x + st];
    }
    __syncthreads();
  }
  if (threadIdx.x == 0) {
    float m = s1[0] / (float)L_;
    float var = (s2[0] - (float)L_ * m * m) / (float)(L_ - 1);
    mean[bf] = m;
    stde[bf] = sqrtf(fmaxf(var, 0.f)) + 1e-5f;
  }
}

// ---------------- Kernel 2: weight f16 conversion ------------------------------
__global__ void k_prep(const float* __restrict__ Wq, const float* __restrict__ Wk,
                       const float* __restrict__ Wv, const float* __restrict__ Wo,
                       const float* __restrict__ Wlin, _Float16* __restrict__ w16,
                       _Float16* __restrict__ Wl16) {
  int i = blockIdx.x * 256 + threadIdx.x;
  if (i < 4096) {
    w16[i]          = (_Float16)Wq[i];
    w16[4096 + i]   = (_Float16)Wk[i];
    w16[8192 + i]   = (_Float16)Wv[i];
    w16[12288 + i]  = (_Float16)Wo[i];
  }
  if (i < P_ * L_) Wl16[i] = (_Float16)Wlin[i];
}

// ---------------- Kernel 3: K,V projections + partial KV / Z -------------------
struct KvScratch {
  float    xn[16 * 64];
  _Float16 h[16 * 64];
  float    Kt[16 * 64];
  float    Vt[16 * 64];
};

__global__ __launch_bounds__(128) void k_kv(
    const float* __restrict__ x, const float* __restrict__ gamma,
    const float* __restrict__ beta, const float* __restrict__ lnw,
    const float* __restrict__ lnb, const _Float16* __restrict__ w16,
    const float* __restrict__ bk, const float* __restrict__ bv,
    const float* __restrict__ mean, const float* __restrict__ stde,
    float* __restrict__ kvpart, float* __restrict__ zpart) {
  __shared__ KvScratch lds[4];
  int wave = threadIdx.x >> 5, lane = threadIdx.x & 31;
  int g = lane >> 4, mn = lane & 15;
  int b = blockIdx.x >> 3;                       // 8 blocks per batch
  int chunk = ((blockIdx.x & 7) << 2) | wave;    // 0..31 : 256 tokens each
  KvScratch& T = lds[wave];
  const _Float16* Wk16 = w16 + 4096;
  const _Float16* Wv16 = w16 + 8192;

  float kvacc[16];
#pragma unroll
  for (int j = 0; j < 16; ++j) kvacc[j] = 0.f;
  float zacc0 = 0.f, zacc1 = 0.f;

  for (int tile = 0; tile < 16; ++tile) {
    int l0 = chunk * 256 + tile * 16;
    // prefetch next tile rows (global_prefetch_b8)
    if (tile + 1 < 16) {
      __builtin_prefetch(&x[((size_t)(b * 64 + lane)) * L_ + l0 + 16], 0, 3);
      __builtin_prefetch(&x[((size_t)(b * 64 + 32 + lane)) * L_ + l0 + 16], 0, 3);
    }
    // ---- load x tile, RevIN normalize into LDS (t,f) -------------------------
#pragma unroll 4
    for (int j = 0; j < 32; ++j) {
      int f = (j << 1) | g;
      float v = x[((size_t)(b * 64 + f)) * L_ + l0 + mn];
      float xnv = (v - mean[b * 64 + f]) / stde[b * 64 + f] * gamma[f] + beta[f];
      T.xn[mn * 64 + f] = xnv;
    }
    // ---- LayerNorm over F per token -----------------------------------------
    float s = 0.f, q = 0.f;
    for (int f = g * 32; f < g * 32 + 32; ++f) {
      float v = T.xn[mn * 64 + f];
      s += v; q += v * v;
    }
    s += __shfl_xor(s, 16, 32);
    q += __shfl_xor(q, 16, 32);
    float mu = s * (1.f / 64.f);
    float rs = rsqrtf(q * (1.f / 64.f) - mu * mu + 1e-5f);
#pragma unroll 4
    for (int j = 0; j < 32; ++j) {
      int t = j >> 1;
      int f = ((j & 1) << 5) | lane;
      float mu_t = __shfl(mu, t, 32);
      float rs_t = __shfl(rs, t, 32);
      float hv = (T.xn[t * 64 + f] - mu_t) * rs_t * lnw[f] + lnb[f];
      T.h[t * 64 + f] = (_Float16)hv;
    }
    // ---- K,V = h @ W^T + b via WMMA -----------------------------------------
    v16h a0 = loadA16x32(T.h, 64, lane, 0);
    v16h a1 = loadA16x32(T.h, 64, lane, 32);
#pragma unroll
    for (int nt = 0; nt < 4; ++nt) {
      int nb = nt * 16;
      float bkn = bk[nb + mn];
      float bvn = bv[nb + mn];
      v8f ck, cv;
#pragma unroll
      for (int r = 0; r < 8; ++r) { ck[r] = bkn; cv[r] = bvn; }
      ck = wmma16(a0, loadB_WT(Wk16, lane, 0, nb), ck);
      ck = wmma16(a1, loadB_WT(Wk16, lane, 32, nb), ck);
      cv = wmma16(a0, loadB_WT(Wv16, lane, 0, nb), cv);
      cv = wmma16(a1, loadB_WT(Wv16, lane, 32, nb), cv);
#pragma unroll
      for (int r = 0; r < 8; ++r) {
        int m = g * 8 + r;
        T.Kt[m * 64 + nb + mn] = phi_f(ck[r]);   // phi(K)
        T.Vt[m * 64 + nb + mn] = cv[r];
      }
    }
    // ---- partial KV(h,d,e) and Z(h,d) over the 16 tokens --------------------
#pragma unroll
    for (int j = 0; j < 16; ++j) {
      int idx = j * 32 + lane;
      int hh = idx >> 6, d = (idx >> 3) & 7, e = idx & 7;
      float acc = 0.f;
#pragma unroll
      for (int t = 0; t < 16; ++t)
        acc += T.Kt[t * 64 + hh * 8 + d] * T.Vt[t * 64 + hh * 8 + e];
      kvacc[j] += acc;
    }
    {
      int hh = lane >> 3, d = lane & 7;
      float a = 0.f, bsum = 0.f;
#pragma unroll
      for (int t = 0; t < 16; ++t) a += T.Kt[t * 64 + hh * 8 + d];
      int idx2 = 32 + lane;
      int h2 = idx2 >> 3, d2 = idx2 & 7;
#pragma unroll
      for (int t = 0; t < 16; ++t) bsum += T.Kt[t * 64 + h2 * 8 + d2];
      zacc0 += a; zacc1 += bsum;
    }
  }
  // deterministic two-stage reduction: write per-wave partials
  size_t wgi = (size_t)b * 32 + chunk;
#pragma unroll
  for (int j = 0; j < 16; ++j) kvpart[wgi * 512 + j * 32 + lane] = kvacc[j];
  zpart[wgi * 64 + lane] = zacc0;
  zpart[wgi * 64 + 32 + lane] = zacc1;
}

// ---------------- Kernel 4: reduce KV/Z partials -------------------------------
__global__ void k_kvred(const float* __restrict__ kvpart,
                        const float* __restrict__ zpart, float* __restrict__ KV,
                        float* __restrict__ Zk) {
  int i = blockIdx.x * 256 + threadIdx.x;
  if (i < B_ * 512) {
    int b = i >> 9, idx = i & 511;
    float s = 0.f;
    for (int c = 0; c < 32; ++c) s += kvpart[((size_t)b * 32 + c) * 512 + idx];
    KV[i] = s;
  }
  if (i < B_ * 64) {
    int b = i >> 6, idx = i & 63;
    float s = 0.f;
    for (int c = 0; c < 32; ++c) s += zpart[((size_t)b * 32 + c) * 64 + idx];
    Zk[i] = s;
  }
}

// ---------------- Kernel 5: Q, attention out, Wo, residual -> y16 --------------
struct AtScratch {
  float    xn[16 * 64];
  _Float16 h[16 * 64];
  _Float16 pq[16 * 64];
  _Float16 att[16 * 64];
  float    nrm[16 * 8];
};

__global__ __launch_bounds__(128) void k_attn(
    const float* __restrict__ x, const float* __restrict__ gamma,
    const float* __restrict__ beta, const float* __restrict__ lnw,
    const float* __restrict__ lnb, const _Float16* __restrict__ w16,
    const float* __restrict__ bq, const float* __restrict__ bo,
    const float* __restrict__ mean, const float* __restrict__ stde,
    const float* __restrict__ KV, const float* __restrict__ Zk,
    _Float16* __restrict__ y16) {
  __shared__ _Float16 Bblk[64 * 64];   // block-diagonal KV as 64x64
  __shared__ float Zs[64];
  __shared__ AtScratch lds[4];
  int wave = threadIdx.x >> 5, lane = threadIdx.x & 31;
  int g = lane >> 4, mn = lane & 15;
  int b = blockIdx.x >> 3;
  int chunk = ((blockIdx.x & 7) << 2) | wave;
  AtScratch& T = lds[wave];
  const _Float16* Wq16 = w16;
  const _Float16* Wo16 = w16 + 12288;

  for (int i = threadIdx.x; i < 4096; i += 128) {
    int k = i >> 6, n = i & 63;
    float v = ((k >> 3) == (n >> 3))
                  ? KV[(size_t)b * 512 + (k >> 3) * 64 + (k & 7) * 8 + (n & 7)]
                  : 0.f;
    Bblk[k * 64 + n] = (_Float16)v;
  }
  if (threadIdx.x < 64) Zs[threadIdx.x] = Zk[b * 64 + threadIdx.x];
  __syncthreads();

  for (int tile = 0; tile < 16; ++tile) {
    int l0 = chunk * 256 + tile * 16;
    if (tile + 1 < 16) {
      __builtin_prefetch(&x[((size_t)(b * 64 + lane)) * L_ + l0 + 16], 0, 3);
      __builtin_prefetch(&x[((size_t)(b * 64 + 32 + lane)) * L_ + l0 + 16], 0, 3);
    }
    // RevIN + residual tile
#pragma unroll 4
    for (int j = 0; j < 32; ++j) {
      int f = (j << 1) | g;
      float v = x[((size_t)(b * 64 + f)) * L_ + l0 + mn];
      float xnv = (v - mean[b * 64 + f]) / stde[b * 64 + f] * gamma[f] + beta[f];
      T.xn[mn * 64 + f] = xnv;
    }
    // LayerNorm
    float s = 0.f, q = 0.f;
    for (int f = g * 32; f < g * 32 + 32; ++f) {
      float v = T.xn[mn * 64 + f];
      s += v; q += v * v;
    }
    s += __shfl_xor(s, 16, 32);
    q += __shfl_xor(q, 16, 32);
    float mu = s * (1.f / 64.f);
    float rs = rsqrtf(q * (1.f / 64.f) - mu * mu + 1e-5f);
#pragma unroll 4
    for (int j = 0; j < 32; ++j) {
      int t = j >> 1;
      int f = ((j & 1) << 5) | lane;
      float mu_t = __shfl(mu, t, 32);
      float rs_t = __shfl(rs, t, 32);
      float hv = (T.xn[t * 64 + f] - mu_t) * rs_t * lnw[f] + lnb[f];
      T.h[t * 64 + f] = (_Float16)hv;
    }
    // Q projection + phi
    v16h a0 = loadA16x32(T.h, 64, lane, 0);
    v16h a1 = loadA16x32(T.h, 64, lane, 32);
#pragma unroll
    for (int nt = 0; nt < 4; ++nt) {
      int nb = nt * 16;
      float bqn = bq[nb + mn];
      v8f cq;
#pragma unroll
      for (int r = 0; r < 8; ++r) cq[r] = bqn;
      cq = wmma16(a0, loadB_WT(Wq16, lane, 0, nb), cq);
      cq = wmma16(a1, loadB_WT(Wq16, lane, 32, nb), cq);
#pragma unroll
      for (int r = 0; r < 8; ++r) {
        int m = g * 8 + r;
        T.pq[m * 64 + nb + mn] = (_Float16)phi_f(cq[r]);
      }
    }
    // normalizer: phiQ . Z per (token, head)
#pragma unroll
    for (int c = 0; c < 4; ++c) {
      int idx = lane * 4 + c;  // 0..127
      int t = idx >> 3, hh = idx & 7;
      float acc = 0.f;
#pragma unroll
      for (int d = 0; d < 8; ++d)
        acc += (float)T.pq[t * 64 + hh * 8 + d] * Zs[hh * 8 + d];
      T.nrm[idx] = acc + 1e-6f;
    }
    // attention out = phiQ @ blockdiag(KV)  (WMMA), then divide by normalizer
    v16h qa0 = loadA16x32(T.pq, 64, lane, 0);
    v16h qa1 = loadA16x32(T.pq, 64, lane, 32);
#pragma unroll
    for (int nt = 0; nt < 4; ++nt) {
      int nb = nt * 16;
      v8f co;
#pragma unroll
      for (int r = 0; r < 8; ++r) co[r] = 0.f;
      co = wmma16(qa0, loadB_RM(Bblk, 64, lane, 0, nb), co);
      co = wmma16(qa1, loadB_RM(Bblk, 64, lane, 32, nb), co);
      int hh = (nb + mn) >> 3;
#pragma unroll
      for (int r = 0; r < 8; ++r) {
        int m = g * 8 + r;
        T.att[m * 64 + nb + mn] = (_Float16)(co[r] / T.nrm[m * 8 + hh]);
      }
    }
    // y = att @ Wo^T + bo + residual -> global f16
    v16h oa0 = loadA16x32(T.att, 64, lane, 0);
    v16h oa1 = loadA16x32(T.att, 64, lane, 32);
#pragma unroll
    for (int nt = 0; nt < 4; ++nt) {
      int nb = nt * 16;
      float bon = bo[nb + mn];
      v8f cy;
#pragma unroll
      for (int r = 0; r < 8; ++r) cy[r] = bon;
      cy = wmma16(oa0, loadB_WT(Wo16, lane, 0, nb), cy);
      cy = wmma16(oa1, loadB_WT(Wo16, lane, 32, nb), cy);
#pragma unroll
      for (int r = 0; r < 8; ++r) {
        int m = g * 8 + r;
        float v = cy[r] + T.xn[m * 64 + nb + mn];
        y16[((size_t)b * L_ + l0 + m) * 64 + nb + mn] = (_Float16)v;
      }
    }
  }
}

// ---------------- Kernel 6: temporal linear Wlin @ y ---------------------------
// K=8192 WMMA reduction; y16 B-tiles staged LDS-double-buffered with CDNA5
// global_load_async_to_lds_b128 (ASYNCcnt), overlapped with compute.
__global__ __launch_bounds__(128) void k_temporal(
    const _Float16* __restrict__ Wl16, const _Float16* __restrict__ y16,
    const float* __restrict__ blin, float* __restrict__ Ztmp) {
  __shared__ _Float16 ybuf[4][2][32 * 16];   // per-wave double buffer, 1KB each
  int wave = threadIdx.x >> 5, lane = threadIdx.x & 31;
  int g = lane >> 4, mn = lane & 15;
  int b = blockIdx.x / 6;
  int t = (blockIdx.x % 6) * 4 + wave;  // 0..23
  int pb = (t >> 2) * 16;
  int fb = (t & 3) * 16;
  v8f acc;
#pragma unroll
  for (int r = 0; r < 8; ++r) acc[r] = 0.f;
  const _Float16* Arow = Wl16 + (size_t)pb * L_;
  const _Float16* Bcol = y16 + (size_t)b * L_ * 64 + fb;

  int rrow = lane >> 1;          // 0..15
  int rhalf = (lane & 1) * 8;    // f16 offset within 16-wide row

  // stage tile kk (32 rows x 16 f16 = 1KB) into ybuf[wave][buf] asynchronously
  auto issue_tile = [&](int kk, int buf) {
#pragma unroll
    for (int s2 = 0; s2 < 2; ++s2) {
      int r = s2 * 16 + rrow;
      unsigned long long ga =
          (unsigned long long)(size_t)(Bcol + (size_t)(kk * 32 + r) * 64 + rhalf);
      unsigned ldsa = (unsigned)(size_t)(&ybuf[wave][buf][r * 16 + rhalf]);
      asm volatile("global_load_async_to_lds_b128 %0, %1, off"
                   :: "v"(ldsa), "v"(ga) : "memory");
    }
  };

  issue_tile(0, 0);
  for (int kk = 0; kk < 256; ++kk) {
    int cur = kk & 1;
    if (kk + 1 < 256) {
      issue_tile(kk + 1, cur ^ 1);
      // 4 outstanding max; <=2 left means current tile's 2 loads completed
      asm volatile("s_wait_asynccnt 2" ::: "memory");
    } else {
      asm volatile("s_wait_asynccnt 0" ::: "memory");
    }
    v16h a = loadA16x32(Arow, L_, lane, kk * 32);
    v16h bf = loadB_RM(&ybuf[wave][cur][0], 16, lane, 0, 0);
    acc = wmma16(a, bf, acc);
  }
#pragma unroll
  for (int r = 0; r < 8; ++r) {
    int m = g * 8 + r;
    Ztmp[((size_t)b * P_ + pb + m) * 64 + fb + mn] = acc[r] + blin[pb + m];
  }
}

// ---------------- Kernel 7: RevIN denorm + projector ---------------------------
__global__ void k_final(const float* __restrict__ Ztmp,
                        const float* __restrict__ gamma,
                        const float* __restrict__ beta,
                        const float* __restrict__ mean,
                        const float* __restrict__ stde,
                        const float* __restrict__ Wp, const float* __restrict__ bp,
                        float* __restrict__ out) {
  int idx = blockIdx.x * 256 + threadIdx.x;
  if (idx >= B_ * P_) return;
  int b = idx / P_, p = idx % P_;
  float acc = bp[0];
  for (int f = 0; f < 64; ++f) {
    float z = Ztmp[((size_t)b * P_ + p) * 64 + f];
    float y = (z - beta[f]) / gamma[f];
    y = y * stde[b * 64 + f] + mean[b * 64 + f];
    acc += Wp[f] * y;
  }
  out[idx] = acc;
}

// ---------------- launch -------------------------------------------------------
extern "C" void kernel_launch(void* const* d_in, const int* in_sizes, int n_in,
                              void* d_out, int out_size, void* d_ws, size_t ws_size,
                              hipStream_t stream) {
  const float* x     = (const float*)d_in[0];
  const float* gamma = (const float*)d_in[1];
  const float* beta  = (const float*)d_in[2];
  const float* lnw   = (const float*)d_in[3];
  const float* lnb   = (const float*)d_in[4];
  const float* Wq    = (const float*)d_in[5];
  const float* bq    = (const float*)d_in[6];
  const float* Wk    = (const float*)d_in[7];
  const float* bk    = (const float*)d_in[8];
  const float* Wv    = (const float*)d_in[9];
  const float* bv    = (const float*)d_in[10];
  const float* Wo    = (const float*)d_in[11];
  const float* bo    = (const float*)d_in[12];
  const float* Wlin  = (const float*)d_in[13];
  const float* blin  = (const float*)d_in[14];
  const float* Wp    = (const float*)d_in[15];
  const float* bp    = (const float*)d_in[16];
  float* out = (float*)d_out;

  // workspace layout
  float* mean   = (float*)d_ws;                     // B*F
  float* stde   = mean + B_ * F_;                   // B*F
  float* KV     = stde + B_ * F_;                   // B*H*8*8 = 16384
  float* Zk     = KV + B_ * 512;                    // B*64
  float* Ztmp   = Zk + B_ * 64;                     // B*P*F = 196608
  float* kvpart = Ztmp + B_ * P_ * F_;              // B*32*512
  float* zpart  = kvpart + (size_t)B_ * 32 * 512;   // B*32*64
  _Float16* w16  = (_Float16*)(zpart + (size_t)B_ * 32 * 64);  // 4*4096
  _Float16* Wl16 = w16 + 4 * 4096;                  // P*L
  _Float16* y16  = Wl16 + (size_t)P_ * L_;          // B*L*F

  k_stats<<<B_ * F_, 256, 0, stream>>>(x, mean, stde);
  k_prep<<<(P_ * L_ + 255) / 256, 256, 0, stream>>>(Wq, Wk, Wv, Wo, Wlin, w16, Wl16);
  k_kv<<<B_ * 8, 128, 0, stream>>>(x, gamma, beta, lnw, lnb, w16, bk, bv, mean,
                                   stde, kvpart, zpart);
  k_kvred<<<(B_ * 512 + 255) / 256, 256, 0, stream>>>(kvpart, zpart, KV, Zk);
  k_attn<<<B_ * 8, 128, 0, stream>>>(x, gamma, beta, lnw, lnb, w16, bq, bo, mean,
                                     stde, KV, Zk, y16);
  k_temporal<<<B_ * 6, 128, 0, stream>>>(Wl16, y16, blin, Ztmp);
  k_final<<<(B_ * P_ + 255) / 256, 256, 0, stream>>>(Ztmp, gamma, beta, mean, stde,
                                                     Wp, bp, out);
}
